// SuperWarpMS_35407710388361
// MI455X (gfx1250) — matrix operand order repaired and checked by reading, not compiled, not run on hardware
//
#include <hip/hip_runtime.h>

// ---------------------------------------------------------------------------
// Types
// ---------------------------------------------------------------------------
typedef _Float16 h16;
typedef __attribute__((ext_vector_type(16))) _Float16 v16h;
typedef __attribute__((ext_vector_type(8)))  _Float16 v8h;
typedef __attribute__((ext_vector_type(8)))  float    v8f;

union ABfrag { v16h v; v8h h8[2]; };
union Pack8  { v8h  v; h16 e[8]; };

#define KPAD_MAX 576   // Cin=64 -> K=576 (exact multiple of 32)
#define FILLV   (-255.0f)

// ---------------------------------------------------------------------------
// Weight pack: (O,I,3,3) f32 -> WMMA B fragments, f16.
// Layout: entry idx = (nt*ksteps + ks)*32 + lane, 16 halves per entry.
// B 32x16 striping: lanes 0..15 hold K = kb+0..15, lanes 16..31 hold
// K = kb+16..31; column n = nt*16 + lane%16. Element e -> K = kb + e.
// ntiles_pad is rounded up to a multiple of 4; padded tiles are zero so the
// conv K-loop needs no guards (EXEC stays all-ones for WMMA).
// ---------------------------------------------------------------------------
__global__ void pack_w_kernel(const float* __restrict__ w, h16* __restrict__ pack,
                              int Cin, int Cout, int ntiles_pad, int ksteps) {
    int idx = blockIdx.x * blockDim.x + threadIdx.x;
    int total = ntiles_pad * ksteps * 32;
    if (idx >= total) return;
    int lane = idx & 31;
    int ks   = (idx >> 5) % ksteps;
    int nt   = (idx >> 5) / ksteps;
    int K = Cin * 9;
    int n = nt * 16 + (lane & 15);
    int kb = ks * 32 + ((lane < 16) ? 0 : 16);
    h16* dst = pack + (size_t)idx * 16;
    for (int e = 0; e < 16; ++e) {
        int k = kb + e;
        h16 v = (h16)0.0f;
        if (k < K && n < Cout) {
            int ci = k / 9, rem = k - ci * 9;
            v = (h16)w[((size_t)(n * Cin + ci) * 3 + rem / 3) * 3 + (rem % 3)];
        }
        dst[e] = v;
    }
}

// ---------------------------------------------------------------------------
// Implicit-GEMM 3x3 conv (pad=1) with WMMA f32_16x16x32_f16, 2x2 register
// blocking per wave.
// Block: 128 threads = 4 waves. Block tile: 64 pixels on row y x 64 output
// channels (4 N-tiles). Wave w -> pixel-group pair (w&1), N-tile pair (w>>1);
// each wave owns 4 accumulators, so every A/B fragment fetched from
// LDS/global feeds 2 WMMAs: 1 ds_load_b128 + 1 global_load_b128 per WMMA.
// Epilogue fuses bias / relu / residual-add / per-pixel mask, b128 stores.
// ---------------------------------------------------------------------------
__global__ __launch_bounds__(128)
void conv3x3_wmma_kernel(const void* __restrict__ inp, int in_is_f32,
                         const h16* __restrict__ wpack,
                         const float* __restrict__ bias,
                         h16* __restrict__ outp,
                         const h16* __restrict__ resid,
                         const h16* __restrict__ maskp,
                         int relu, int H, int W, int Cin, int Cout,
                         int ksteps) {
    __shared__ h16 lds_a[64 * KPAD_MAX];

    const int Kpad = ksteps * 32;
    const int K    = Cin * 9;
    const int x0   = blockIdx.x * 64;
    const int y    = blockIdx.y;
    const int ntb  = blockIdx.z * 4;
    const int tid  = threadIdx.x;

    const float* inF = (const float*)inp;
    const h16*   inH = (const h16*)inp;

    // -------- cooperative im2col fill (zero-padded borders) --------
    const int tot = 64 * Kpad;
    for (int e = tid; e < tot; e += 128) {
        int p = e / Kpad;
        int k = e - p * Kpad;
        h16 v = (h16)0.0f;
        if (k < K) {
            int ci  = k / 9;
            int rem = k - ci * 9;
            int iy  = y  + rem / 3 - 1;
            int ix  = x0 + p + (rem % 3) - 1;
            if (iy >= 0 && iy < H && ix >= 0 && ix < W) {
                size_t off = ((size_t)ci * H + iy) * W + ix;
                v = in_is_f32 ? (h16)inF[off] : inH[off];
            }
        }
        lds_a[e] = v;
    }
    __syncthreads();

    const int wave = tid >> 5;
    const int lane = tid & 31;
    const int pg0  = (wave & 1) * 2;        // pixel groups pg0, pg0+1
    const int nt0  = ntb + (wave >> 1) * 2; // N-tiles nt0, nt0+1

    // A fragment rows: lane holds row m = lane%16; halves 0..15 -> K=kb+lsel+e
    // (contiguous order, matching the B pack).
    const int  lsel  = (lane < 16) ? 0 : 16;
    const h16* arow0 = lds_a + (size_t)(pg0 * 16 + (lane & 15)) * Kpad;
    const h16* arow1 = arow0 + (size_t)16 * Kpad;
    const h16* wl0   = wpack + ((size_t)(nt0 * ksteps) * 32 + lane) * 16;
    const h16* wl1   = wl0 + (size_t)ksteps * 32 * 16;

    v8f acc00 = {}, acc01 = {}, acc10 = {}, acc11 = {};
    for (int ks = 0; ks < ksteps; ++ks) {
        int kb = ks * 32 + lsel;
        ABfrag a0, a1, b0, b1;
        a0.h8[0] = *(const v8h*)(arow0 + kb);       // ds_load_b128
        a0.h8[1] = *(const v8h*)(arow0 + kb + 8);
        a1.h8[0] = *(const v8h*)(arow1 + kb);
        a1.h8[1] = *(const v8h*)(arow1 + kb + 8);
        b0.h8[0] = *(const v8h*)(wl0);              // global_load_b128
        b0.h8[1] = *(const v8h*)(wl0 + 8);
        b1.h8[0] = *(const v8h*)(wl1);
        b1.h8[1] = *(const v8h*)(wl1 + 8);
        wl0 += 32 * 16;
        wl1 += 32 * 16;
        __builtin_prefetch(wl0, 0, 1);              // global_prefetch_b8
        __builtin_prefetch(wl1, 0, 1);
        acc00 = __builtin_amdgcn_wmma_f32_16x16x32_f16(
                    false, a0.v, false, b0.v, (short)0, acc00, false, false);
        acc01 = __builtin_amdgcn_wmma_f32_16x16x32_f16(
                    false, a0.v, false, b1.v, (short)0, acc01, false, false);
        acc10 = __builtin_amdgcn_wmma_f32_16x16x32_f16(
                    false, a1.v, false, b0.v, (short)0, acc10, false, false);
        acc11 = __builtin_amdgcn_wmma_f32_16x16x32_f16(
                    false, a1.v, false, b1.v, (short)0, acc11, false, false);
    }

    // -------- epilogue: 4 output tiles --------
    const int mb = (lane < 16) ? 0 : 8;     // D: VGPR v -> row m = mb+v
    v8f accs[4] = { acc00, acc01, acc10, acc11 };
    #pragma unroll
    for (int t = 0; t < 4; ++t) {
        int pg = pg0 + (t >> 1);
        int nt = nt0 + (t & 1);
        int n  = nt * 16 + (lane & 15);
        if (n < Cout) {
            int xb = x0 + pg * 16 + mb;     // 8 consecutive x positions
            float bi = bias[n];
            Pack8 o;
            #pragma unroll
            for (int v = 0; v < 8; ++v) {
                int xx = xb + v;
                float val = accs[t][v] + bi;
                if (relu) val = val > 0.0f ? val : 0.0f;
                if (resid) val += (float)resid[((size_t)n * H + y) * W + xx];
                if (maskp) val *= (float)maskp[(size_t)y * W + xx];
                o.e[v] = (h16)val;
            }
            *(v8h*)(outp + ((size_t)n * H + y) * W + xb) = o.v;  // b128 store
        }
    }
}

// ---------------------------------------------------------------------------
// Pixel shuffle r=2: in (4C,H,W) f16 -> out (C,2H,2W) f16
// ---------------------------------------------------------------------------
__global__ void pixshuf_kernel(const h16* __restrict__ in, h16* __restrict__ out,
                               int C, int H, int W) {
    size_t i = (size_t)blockIdx.x * blockDim.x + threadIdx.x;
    size_t total = (size_t)C * 4 * H * W;
    if (i >= total) return;
    int OW = 2 * W, OH = 2 * H;
    int c  = (int)(i / ((size_t)OH * OW));
    int r  = (int)(i - (size_t)c * OH * OW);
    int oy = r / OW, ox = r - oy * OW;
    int ry = oy & 1, rx = ox & 1;
    out[i] = in[((size_t)(c * 4 + ry * 2 + rx) * H + (oy >> 1)) * W + (ox >> 1)];
}

// ---------------------------------------------------------------------------
// Homography setup: minv = inverse(m @ diag(s,s,1))   (single thread)
// ---------------------------------------------------------------------------
__global__ void warp_setup_kernel(const float* __restrict__ m, float* __restrict__ minv,
                                  float s) {
    if (threadIdx.x != 0 || blockIdx.x != 0) return;
    float a[9];
    for (int i = 0; i < 3; ++i)
        for (int j = 0; j < 3; ++j)
            a[i * 3 + j] = m[i * 3 + j] * ((j < 2) ? s : 1.0f);
    float c00 =  a[4] * a[8] - a[5] * a[7];
    float c01 = -(a[3] * a[8] - a[5] * a[6]);
    float c02 =  a[3] * a[7] - a[4] * a[6];
    float det = a[0] * c00 + a[1] * c01 + a[2] * c02;
    float id  = 1.0f / det;
    minv[0] = c00 * id;
    minv[1] = -(a[1] * a[8] - a[2] * a[7]) * id;
    minv[2] =  (a[1] * a[5] - a[2] * a[4]) * id;
    minv[3] = c01 * id;
    minv[4] =  (a[0] * a[8] - a[2] * a[6]) * id;
    minv[5] = -(a[0] * a[5] - a[2] * a[3]) * id;
    minv[6] = c02 * id;
    minv[7] = -(a[0] * a[7] - a[1] * a[6]) * id;
    minv[8] =  (a[0] * a[4] - a[1] * a[3]) * id;
}

__device__ __forceinline__ float cubw(float d) {
    const float a = -0.5f;
    float d2 = d * d, d3 = d2 * d;
    return (d <= 1.0f) ? ((a + 2.0f) * d3 - (a + 3.0f) * d2 + 1.0f)
                       : (a * d3 - 5.0f * a * d2 + 8.0f * a * d - 4.0f * a);
}

// ---------------------------------------------------------------------------
// Bicubic homography warp of the 3x768x768 f16 image to 896x896.
// Writes: base (f32, y/3 with FILL), mask (f16 0/1), r0in (f16, mask * y/3).
// ---------------------------------------------------------------------------
__global__ void bicubic_warp_kernel(const h16* __restrict__ p4,
                                    const float* __restrict__ minv,
                                    float* __restrict__ base,
                                    h16* __restrict__ mask,
                                    h16* __restrict__ r0in,
                                    int Hs, int Ws, int OH, int OW) {
    int idx = blockIdx.x * blockDim.x + threadIdx.x;
    if (idx >= OH * OW) return;
    int i = idx / OW, j = idx - i * OW;
    float jf = (float)j, iff = (float)i;
    float X = minv[0] * jf + minv[1] * iff + minv[2];
    float Y = minv[3] * jf + minv[4] * iff + minv[5];
    float Z = minv[6] * jf + minv[7] * iff + minv[8];
    float gx = X / Z, gy = Y / Z;
    float ixf = floorf(gx), iyf = floorf(gy);
    float fx = gx - ixf, fy = gy - iyf;
    int ixi = (int)ixf, iyi = (int)iyf;
    bool valid = (ixi >= 1) && (ixi <= Ws - 3) && (iyi >= 1) && (iyi <= Hs - 3);

    float wx[4] = { cubw(fx + 1.0f), cubw(fx), cubw(1.0f - fx), cubw(2.0f - fx) };
    float wy[4] = { cubw(fy + 1.0f), cubw(fy), cubw(1.0f - fy), cubw(2.0f - fy) };

    float acc[3] = {0.0f, 0.0f, 0.0f};
    size_t plane = (size_t)Hs * Ws;
    #pragma unroll
    for (int ii = 0; ii < 4; ++ii) {
        int yy = iyi - 1 + ii;
        yy = yy < 0 ? 0 : (yy > Hs - 1 ? Hs - 1 : yy);
        float wyv = wy[ii];
        #pragma unroll
        for (int jj = 0; jj < 4; ++jj) {
            int xx = ixi - 1 + jj;
            xx = xx < 0 ? 0 : (xx > Ws - 1 ? Ws - 1 : xx);
            float w = wyv * wx[jj];
            size_t off = (size_t)yy * Ws + xx;
            acc[0] += w * (float)p4[off];
            acc[1] += w * (float)p4[plane + off];
            acc[2] += w * (float)p4[2 * plane + off];
        }
    }
    size_t n = (size_t)OH * OW;
    mask[idx] = (h16)(valid ? 1.0f : 0.0f);
    #pragma unroll
    for (int c = 0; c < 3; ++c) {
        float yv = valid ? acc[c] : FILLV;
        float bv = yv * (1.0f / 3.0f);
        base[(size_t)c * n + idx] = bv;
        r0in[(size_t)c * n + idx] = (h16)(valid ? bv : 0.0f);
    }
}

// ---------------------------------------------------------------------------
// Final combine: out = mask ? (base + h) : FILL   (f32 output)
// ---------------------------------------------------------------------------
__global__ void finalize_kernel(const float* __restrict__ base,
                                const h16* __restrict__ mask,
                                const h16* __restrict__ r3,
                                float* __restrict__ out, int npix) {
    size_t idx = (size_t)blockIdx.x * blockDim.x + threadIdx.x;
    if (idx >= (size_t)3 * npix) return;
    size_t pix = idx % npix;
    float m = (float)mask[pix];
    out[idx] = (m > 0.5f) ? (base[idx] + (float)r3[idx]) : FILLV;
}

// ---------------------------------------------------------------------------
// Host orchestration
// ---------------------------------------------------------------------------
extern "C" void kernel_launch(void* const* d_in, const int* in_sizes, int n_in,
                              void* d_out, int out_size, void* d_ws, size_t ws_size,
                              hipStream_t stream) {
    (void)in_sizes; (void)n_in; (void)out_size; (void)ws_size;

    const int LR = 192, MD = 384, HR = 768, OUT = 896;
    char* ws = (char*)d_ws;
    size_t off = 0;
    auto alloc = [&](size_t bytes) -> char* {
        off = (off + 255) & ~(size_t)255;
        char* p = ws + off;
        off += bytes;
        return p;
    };

    // ---- layer table (param indices into d_in; dict order) ----
    int widx[25], bidx[25], cins[25], couts[25];
    int li = 0;
    widx[li] = 4;  bidx[li] = 5;  cins[li] = 3;  couts[li] = 64;  li++;   // head
    for (int i = 0; i < 8; ++i) {
        widx[li] = 6 + 4 * i; bidx[li] = 7 + 4 * i; cins[li] = 64; couts[li] = 64; li++; // res w1
        widx[li] = 8 + 4 * i; bidx[li] = 9 + 4 * i; cins[li] = 64; couts[li] = 64; li++; // res w2
    }
    widx[li] = 38; bidx[li] = 39; cins[li] = 64; couts[li] = 64;  li++;   // body   (17)
    widx[li] = 40; bidx[li] = 41; cins[li] = 64; couts[li] = 256; li++;   // up0    (18)
    widx[li] = 42; bidx[li] = 43; cins[li] = 64; couts[li] = 256; li++;   // up1    (19)
    widx[li] = 44; bidx[li] = 45; cins[li] = 64; couts[li] = 3;   li++;   // tail   (20)
    widx[li] = 46; bidx[li] = 47; cins[li] = 3;  couts[li] = 64;  li++;   // r0     (21)
    widx[li] = 48; bidx[li] = 49; cins[li] = 64; couts[li] = 64;  li++;   // r1     (22)
    widx[li] = 50; bidx[li] = 51; cins[li] = 64; couts[li] = 64;  li++;   // r2     (23)
    widx[li] = 52; bidx[li] = 53; cins[li] = 64; couts[li] = 3;   li++;   // r3     (24)

    auto nt_pad = [](int cout) { return (((cout + 15) / 16) + 3) & ~3; };

    // ---- workspace layout ----
    float* MINV = (float*)alloc(16 * sizeof(float));
    h16* packp[25];
    for (int l = 0; l < 25; ++l) {
        int ksteps = (cins[l] * 9 + 31) / 32;
        packp[l] = (h16*)alloc((size_t)nt_pad(couts[l]) * ksteps * 32 * 16 * sizeof(h16));
    }
    size_t lr2 = (size_t)LR * LR, md2 = (size_t)MD * MD, hr2 = (size_t)HR * HR;
    size_t ot2 = (size_t)OUT * OUT;
    h16* HB   = (h16*)alloc(64 * lr2 * 2);
    h16* F1b  = (h16*)alloc(64 * lr2 * 2);
    h16* F2b  = (h16*)alloc(64 * lr2 * 2);
    h16* Tb   = (h16*)alloc(64 * lr2 * 2);
    h16* U0   = (h16*)alloc(256 * lr2 * 2);
    h16* S0   = (h16*)alloc(64 * md2 * 2);
    h16* U1   = (h16*)alloc(256 * md2 * 2);
    h16* S1   = (h16*)alloc(64 * hr2 * 2);
    h16* P4   = (h16*)alloc(3 * hr2 * 2);
    float* BASE = (float*)alloc(3 * ot2 * 4);
    h16* MASK = (h16*)alloc(ot2 * 2);
    h16* R0IN = (h16*)alloc(3 * ot2 * 2);
    h16* G0   = (h16*)alloc(64 * ot2 * 2);
    h16* G1   = (h16*)alloc(64 * ot2 * 2);
    h16* RT   = (h16*)alloc(3 * ot2 * 2);

    // ---- pack all weights (zero-padded N-tiles; keeps K-loop guard-free) ----
    for (int l = 0; l < 25; ++l) {
        int ksteps = (cins[l] * 9 + 31) / 32;
        int ntp = nt_pad(couts[l]);
        int tot = ntp * ksteps * 32;
        pack_w_kernel<<<(tot + 255) / 256, 256, 0, stream>>>(
            (const float*)d_in[widx[l]], packp[l], cins[l], couts[l], ntp, ksteps);
    }

    auto conv = [&](const void* in, int inf32, int l, h16* out,
                    const h16* res, const h16* msk, int relu, int H, int W) {
        int ksteps = (cins[l] * 9 + 31) / 32;
        int ntp = nt_pad(couts[l]);
        dim3 g(W / 64, H, ntp / 4);
        conv3x3_wmma_kernel<<<g, 128, 0, stream>>>(
            in, inf32, packp[l], (const float*)d_in[bidx[l]], out, res, msk,
            relu, H, W, cins[l], couts[l], ksteps);
    };

    // ---- EDSR body at 192x192 ----
    conv(d_in[0], 1, 0, HB, nullptr, nullptr, 0, LR, LR);      // head (f32 in)
    const h16* f = HB;
    h16* ping = F1b; h16* pong = F2b;
    int lw = 1;
    for (int i = 0; i < 8; ++i) {
        conv(f,  0, lw++, Tb,   nullptr, nullptr, 1, LR, LR);  // relu(conv1)
        conv(Tb, 0, lw++, ping, f,       nullptr, 0, LR, LR);  // conv2 + f
        f = ping; h16* t = ping; ping = pong; pong = t;
    }
    conv(f, 0, 17, ping, HB, nullptr, 0, LR, LR);              // body + head
    f = ping;

    // ---- upsample x4 ----
    conv(f, 0, 18, U0, nullptr, nullptr, 0, LR, LR);           // 64 -> 256
    {
        size_t tot = 256 * lr2;
        pixshuf_kernel<<<(tot + 255) / 256, 256, 0, stream>>>(U0, S0, 64, LR, LR);
    }
    conv(S0, 0, 19, U1, nullptr, nullptr, 0, MD, MD);          // 64 -> 256
    {
        size_t tot = 256 * md2;
        pixshuf_kernel<<<(tot + 255) / 256, 256, 0, stream>>>(U1, S1, 64, MD, MD);
    }
    conv(S1, 0, 20, P4, nullptr, nullptr, 0, HR, HR);          // tail 64 -> 3

    // ---- bicubic homography warp to 896x896 ----
    warp_setup_kernel<<<1, 32, 0, stream>>>((const float*)d_in[1], MINV,
                                            (float)LR / (float)HR);
    {
        int tot = OUT * OUT;
        bicubic_warp_kernel<<<(tot + 255) / 256, 256, 0, stream>>>(
            P4, MINV, BASE, MASK, R0IN, HR, HR, OUT, OUT);
    }

    // ---- refinement head (mask folded into conv epilogues) ----
    conv(R0IN, 0, 21, G0, nullptr, MASK, 1, OUT, OUT);         // 3  -> 64
    conv(G0,   0, 22, G1, nullptr, MASK, 1, OUT, OUT);         // 64 -> 64
    conv(G1,   0, 23, G0, nullptr, MASK, 1, OUT, OUT);         // 64 -> 64
    conv(G0,   0, 24, RT, nullptr, nullptr, 0, OUT, OUT);      // 64 -> 3

    {
        size_t tot = (size_t)3 * OUT * OUT;
        finalize_kernel<<<(tot + 255) / 256, 256, 0, stream>>>(
            BASE, MASK, RT, (float*)d_out, OUT * OUT);
    }
}
